// DSVTBlock_35579509080406
// MI455X (gfx1250) — matrix operand order repaired and checked by reading, not compiled, not run on hardware
//
#include <hip/hip_runtime.h>

// ---------------------------------------------------------------------------
// DSVT block (2 encoder layers) for MI455X / gfx1250 (wave32, WMMA bf16,
// async global->LDS staging with double buffering)
// ---------------------------------------------------------------------------

typedef __bf16 bf16_t;
typedef __attribute__((ext_vector_type(16))) __bf16 bf16x16;
typedef __attribute__((ext_vector_type(8)))  __bf16 bf16x8;
typedef __attribute__((ext_vector_type(8)))  float  f32x8;

union BF16Frag { bf16x16 v; bf16x8 h[2]; };

static constexpr int    kN  = 65536;
static constexpr int    kC  = 512;
static constexpr int    kH  = 8;
static constexpr int    kL  = 64;
static constexpr int    kS  = 1024;
static constexpr int    kFF = 2048;
static constexpr size_t kNC = (size_t)kN * kC;

// ---- workspace layout (bytes) ----
static constexpr size_t OFF_QK   = 0;                              // bf16 N*C (later aliased as attn-out)
static constexpr size_t OFF_FEAT = OFF_QK   + kNC * 2;             // bf16 N*C (later aliased as x_bf16)
static constexpr size_t OFF_QKV  = OFF_FEAT + kNC * 2;             // bf16 N*3C (later aliased as f32 N*C "u")
static constexpr size_t OFF_TMP  = OFF_QKV  + kNC * 3 * 2;         // f32 N*C (proj out, then ffn2 out)
static constexpr size_t OFF_X    = OFF_TMP  + kNC * 4;             // f32 N*C
static constexpr size_t OFF_FFN1 = OFF_X    + kNC * 4;             // bf16 N*FF
static constexpr size_t OFF_Y    = OFF_FFN1 + (size_t)kN * kFF * 2;// f32 N*C (layer0 output)
static constexpr size_t OFF_W    = OFF_Y    + kNC * 4;             // bf16 weights (~6.3 MB)

__device__ inline f32x8 zero8() {
    f32x8 z;
#pragma unroll
    for (int e = 0; e < 8; ++e) z[e] = 0.0f;
    return z;
}

// Low 32 bits of a generic pointer to LDS = LDS byte offset.
__device__ inline unsigned lds_off(const void* p) {
    return (unsigned)(uintptr_t)p;
}

// Stage 64B of A and 64B of B (per lane) into LDS via async DMA.
// Per the ISA, INST_OFFSET is added to both the LDS and global address,
// so one (ldsaddr, gaddr) pair covers 4 x b128 with offset:0/16/32/48.
__device__ inline void async_stage(const bf16_t* gA, const bf16_t* gB,
                                   unsigned ldsA, unsigned ldsB) {
    unsigned long long a = (unsigned long long)(uintptr_t)gA;
    unsigned long long b = (unsigned long long)(uintptr_t)gB;
    asm volatile(
        "global_load_async_to_lds_b128 %0, %1, off\n\t"
        "global_load_async_to_lds_b128 %0, %1, off offset:16\n\t"
        "global_load_async_to_lds_b128 %0, %1, off offset:32\n\t"
        "global_load_async_to_lds_b128 %0, %1, off offset:48\n\t"
        "global_load_async_to_lds_b128 %2, %3, off\n\t"
        "global_load_async_to_lds_b128 %2, %3, off offset:16\n\t"
        "global_load_async_to_lds_b128 %2, %3, off offset:32\n\t"
        "global_load_async_to_lds_b128 %2, %3, off offset:48\n\t"
        :
        : "v"(ldsA), "v"(a), "v"(ldsB), "v"(b)
        : "memory");
}

// ---------------------------------------------------------------------------
// f32 -> bf16 elementwise
// ---------------------------------------------------------------------------
__global__ void f2bf_kernel(const float* __restrict__ in, bf16_t* __restrict__ out, int n) {
    int i = blockIdx.x * blockDim.x + threadIdx.x;
    if (i < n) out[i] = (bf16_t)in[i];
}

// ---------------------------------------------------------------------------
// gather: row j -> voxel v = inds[j]; feat = src[v], qk = src[v] + pos[v]
// ---------------------------------------------------------------------------
__global__ __launch_bounds__(256)
void gather_kernel(const float* __restrict__ src, const float* __restrict__ pos,
                   const int* __restrict__ inds,
                   bf16_t* __restrict__ qk, bf16_t* __restrict__ feat) {
    int j = blockIdx.x;
    int v = inds[j];
    const float* s = src + (size_t)v * kC;
    const float* p = pos + (size_t)v * kC;
    for (int c = threadIdx.x; c < kC; c += 256) {
        float sv = s[c];
        feat[(size_t)j * kC + c] = (bf16_t)sv;
        qk[(size_t)j * kC + c]   = (bf16_t)(sv + p[c]);
    }
}

// ---------------------------------------------------------------------------
// GEMM: Out[m][n] = sum_k A[m][k]*B[n][k] + bias[n]   (A, B row-major bf16)
// 128 threads (4 waves), 128x128 block tile; wave tile 64x64 = 4x4 WMMAs.
// K staged via async global->LDS DMA in 32-wide slabs, double buffered.
// ---------------------------------------------------------------------------
__global__ __launch_bounds__(128)
void gemm_bf16_kernel(const bf16_t* __restrict__ A, int lda,
                      const bf16_t* __restrict__ B, int ldb,
                      const float* __restrict__ bias,
                      float* __restrict__ Cf, bf16_t* __restrict__ Cb, int ldc,
                      int K, int relu) {
    __shared__ bf16_t As[2][128 * 32];
    __shared__ bf16_t Bs[2][128 * 32];

    const int m0   = blockIdx.y * 128;
    const int n0   = blockIdx.x * 128;
    const int t    = threadIdx.x;
    const int lane = t & 31;
    const int wave = t >> 5;
    const int wm   = (wave >> 1) * 64;
    const int wn   = (wave & 1) * 64;

    f32x8 acc[4][4];
#pragma unroll
    for (int i = 0; i < 4; ++i)
#pragma unroll
        for (int j = 0; j < 4; ++j) acc[i][j] = zero8();

    const int mrow = lane & 15;          // fragment row / col within 16
    const int kh   = (lane >> 4) * 8;    // A-frag K sub-offset
    const int kb   = (lane >> 4) * 16;   // unused (B-frag also uses kh)

    const bf16_t* gA = A + (size_t)(m0 + t) * lda;
    const bf16_t* gB = B + (size_t)(n0 + t) * ldb;
    const unsigned ldsA[2] = { lds_off(&As[0][t * 32]), lds_off(&As[1][t * 32]) };
    const unsigned ldsB[2] = { lds_off(&Bs[0][t * 32]), lds_off(&Bs[1][t * 32]) };

    const int nstages = K >> 5;
    async_stage(gA, gB, ldsA[0], ldsB[0]);          // stage 0 in flight

    for (int st = 0; st < nstages; ++st) {
        const int buf  = st & 1;
        const bool more = (st + 1 < nstages);
        if (more) {
            async_stage(gA + (st + 1) * 32, gB + (st + 1) * 32,
                        ldsA[buf ^ 1], ldsB[buf ^ 1]);
            // 16 async ops outstanding (8 current + 8 next); async loads
            // complete in order, so <=8 means the current stage has landed.
            asm volatile("s_wait_asynccnt 0x8" ::: "memory");
        } else {
            asm volatile("s_wait_asynccnt 0x0" ::: "memory");
        }
        __syncthreads();

        BF16Frag a[4];
#pragma unroll
        for (int i = 0; i < 4; ++i) {
            const bf16_t* p = &As[buf][(wm + i * 16 + mrow) * 32];
            a[i].h[0] = *(const bf16x8*)(p + kh);
            a[i].h[1] = *(const bf16x8*)(p + 16 + kh);
        }
#pragma unroll
        for (int j = 0; j < 4; ++j) {
            BF16Frag b;
            const bf16_t* p = &Bs[buf][(wn + j * 16 + mrow) * 32 + kh];
            b.h[0] = *(const bf16x8*)(p);
            b.h[1] = *(const bf16x8*)(p + 16);
#pragma unroll
            for (int i = 0; i < 4; ++i) {
                acc[i][j] = __builtin_amdgcn_wmma_f32_16x16x32_bf16(
                    false, a[i].v, false, b.v, (short)0, acc[i][j], false, false);
            }
        }
        __syncthreads();   // all waves done reading buf before it is re-staged
    }

    // epilogue: C layout -> VGPR r: M = r + 8*(lane>=16), N = lane&15
    const int nc   = lane & 15;
    const int mofs = (lane >> 4) * 8;
#pragma unroll
    for (int j = 0; j < 4; ++j) {
        int n = n0 + wn + j * 16 + nc;
        float bv = bias ? bias[n] : 0.0f;
#pragma unroll
        for (int i = 0; i < 4; ++i) {
#pragma unroll
            for (int r = 0; r < 8; ++r) {
                int m = m0 + wm + i * 16 + mofs + r;
                float v = acc[i][j][r] + bv;
                if (relu) v = fmaxf(v, 0.0f);
                if (Cf) Cf[(size_t)m * ldc + n] = v;
                else    Cb[(size_t)m * ldc + n] = (bf16_t)v;
            }
        }
    }
}

// ---------------------------------------------------------------------------
// Attention: one wave per (set, head). Q,K,V are 64x64 (L x D).
// scores = (Q K^T) * 1/sqrt(D) -> masked softmax -> attn @ V.
// ---------------------------------------------------------------------------
__global__ __launch_bounds__(32)
void attn_kernel(const bf16_t* __restrict__ qkv,
                 const unsigned char* __restrict__ mask,
                 bf16_t* __restrict__ outb) {
    __shared__ bf16_t Qs[64 * 64];
    __shared__ bf16_t Ks[64 * 64];
    __shared__ bf16_t Vt[64 * 64];   // V transposed: Vt[d][m]
    __shared__ float  Sc[64 * 64];
    __shared__ bf16_t Ab[64 * 64];

    const int s    = blockIdx.x >> 3;
    const int h    = blockIdx.x & 7;
    const int lane = threadIdx.x;
    const size_t rowBase = (size_t)s * kL;

    // ---- stage Q, K, V(transposed) into LDS ----
    for (int chunk = lane; chunk < 512; chunk += 32) {
        int l  = chunk >> 3;
        int dc = (chunk & 7) * 8;
        const bf16_t* base = qkv + (rowBase + l) * (3 * kC) + h * 64 + dc;
        bf16x8 q = *(const bf16x8*)(base);
        bf16x8 k = *(const bf16x8*)(base + kC);
        bf16x8 v = *(const bf16x8*)(base + 2 * kC);
        *(bf16x8*)&Qs[l * 64 + dc] = q;
        *(bf16x8*)&Ks[l * 64 + dc] = k;
#pragma unroll
        for (int e = 0; e < 8; ++e) Vt[(dc + e) * 64 + l] = v[e];
    }
    __syncthreads();

    const int mrow = lane & 15;
    const int kh   = (lane >> 4) * 8;
    const int kb   = (lane >> 4) * 16;
    const int nc   = lane & 15;
    const int mofs = (lane >> 4) * 8;

    // ---- scores = Q K^T ----
    f32x8 acc[4][4];
#pragma unroll
    for (int i = 0; i < 4; ++i)
#pragma unroll
        for (int j = 0; j < 4; ++j) acc[i][j] = zero8();

#pragma unroll
    for (int ks = 0; ks < 2; ++ks) {
        BF16Frag a[4];
#pragma unroll
        for (int i = 0; i < 4; ++i) {
            const bf16_t* p = &Qs[(i * 16 + mrow) * 64 + ks * 32];
            a[i].h[0] = *(const bf16x8*)(p + kh);
            a[i].h[1] = *(const bf16x8*)(p + 16 + kh);
        }
#pragma unroll
        for (int j = 0; j < 4; ++j) {
            BF16Frag b;
            const bf16_t* p = &Ks[(j * 16 + mrow) * 64 + ks * 32 + kh];
            b.h[0] = *(const bf16x8*)(p);
            b.h[1] = *(const bf16x8*)(p + 16);
#pragma unroll
            for (int i = 0; i < 4; ++i)
                acc[i][j] = __builtin_amdgcn_wmma_f32_16x16x32_bf16(
                    false, a[i].v, false, b.v, (short)0, acc[i][j], false, false);
        }
    }

    const float scale = 0.125f;   // 1/sqrt(64)
#pragma unroll
    for (int i = 0; i < 4; ++i)
#pragma unroll
        for (int j = 0; j < 4; ++j)
#pragma unroll
            for (int r = 0; r < 8; ++r)
                Sc[(i * 16 + mofs + r) * 64 + j * 16 + nc] = acc[i][j][r] * scale;
    __syncthreads();

    // ---- softmax (2 rows per lane) ----
    const unsigned char* mrow_mask = mask + (size_t)s * kL;
#pragma unroll
    for (int rr = 0; rr < 2; ++rr) {
        int l = lane + rr * 32;
        float* row = &Sc[l * 64];
        float mx = -1e30f;
        for (int m = 0; m < 64; ++m) {
            float v = row[m];
            if (mrow_mask[m]) v = -1e9f;
            row[m] = v;
            mx = fmaxf(mx, v);
        }
        float sum = 0.0f;
        for (int m = 0; m < 64; ++m) {
            float e = __expf(row[m] - mx);
            row[m] = e;
            sum += e;
        }
        float inv = 1.0f / sum;
        for (int m = 0; m < 64; ++m) Ab[l * 64 + m] = (bf16_t)(row[m] * inv);
    }
    __syncthreads();

    // ---- out = attn @ V  (contraction over keys m; B-frag reads Vt rows) ----
#pragma unroll
    for (int i = 0; i < 4; ++i)
#pragma unroll
        for (int j = 0; j < 4; ++j) acc[i][j] = zero8();

#pragma unroll
    for (int ks = 0; ks < 2; ++ks) {
        BF16Frag a[4];
#pragma unroll
        for (int i = 0; i < 4; ++i) {
            const bf16_t* p = &Ab[(i * 16 + mrow) * 64 + ks * 32];
            a[i].h[0] = *(const bf16x8*)(p + kh);
            a[i].h[1] = *(const bf16x8*)(p + 16 + kh);
        }
#pragma unroll
        for (int j = 0; j < 4; ++j) {
            BF16Frag b;
            const bf16_t* p = &Vt[(j * 16 + mrow) * 64 + ks * 32 + kh];
            b.h[0] = *(const bf16x8*)(p);
            b.h[1] = *(const bf16x8*)(p + 16);
#pragma unroll
            for (int i = 0; i < 4; ++i)
                acc[i][j] = __builtin_amdgcn_wmma_f32_16x16x32_bf16(
                    false, a[i].v, false, b.v, (short)0, acc[i][j], false, false);
        }
    }

#pragma unroll
    for (int i = 0; i < 4; ++i)
#pragma unroll
        for (int j = 0; j < 4; ++j)
#pragma unroll
            for (int r = 0; r < 8; ++r) {
                int l = i * 16 + mofs + r;
                int d = j * 16 + nc;
                outb[(rowBase + l) * kC + h * 64 + d] = (bf16_t)acc[i][j][r];
            }
}

// ---------------------------------------------------------------------------
// Fused add (+optional permutation scatter) + LayerNorm.
//   v = inds ? inds[j] : j ;  t = a[v] + b[j] ;  out[v] = LN(t)*g + beta
// Optional bf16 mirror of the output (FFN GEMM input).
// ---------------------------------------------------------------------------
__global__ __launch_bounds__(256)
void add_ln_kernel(const float* __restrict__ a, const float* __restrict__ b,
                   const int* __restrict__ inds,
                   const float* __restrict__ g, const float* __restrict__ beta,
                   float* __restrict__ outf, bf16_t* __restrict__ outb) {
    int j = blockIdx.x;
    int v = inds ? inds[j] : j;
    const float* pa = a + (size_t)v * kC;
    const float* pb = b + (size_t)j * kC;
    int t = threadIdx.x;

    float vals[2], lsum = 0.0f, lsq = 0.0f;
#pragma unroll
    for (int e = 0; e < 2; ++e) {
        int c = t + e * 256;
        float x = pa[c] + pb[c];
        vals[e] = x;
        lsum += x;
        lsq  += x * x;
    }
    __shared__ float redS[8], redQ[8];
    for (int off = 16; off > 0; off >>= 1) {
        lsum += __shfl_xor(lsum, off, 32);
        lsq  += __shfl_xor(lsq,  off, 32);
    }
    int wid = t >> 5, ln = t & 31;
    if (ln == 0) { redS[wid] = lsum; redQ[wid] = lsq; }
    __syncthreads();
    if (t == 0) {
        float s = 0.0f, q = 0.0f;
        for (int i = 0; i < 8; ++i) { s += redS[i]; q += redQ[i]; }
        redS[0] = s; redQ[0] = q;
    }
    __syncthreads();
    float mean = redS[0] * (1.0f / kC);
    float var  = redQ[0] * (1.0f / kC) - mean * mean;
    float rstd = rsqrtf(var + 1e-5f);
#pragma unroll
    for (int e = 0; e < 2; ++e) {
        int c = t + e * 256;
        float y = (vals[e] - mean) * rstd * g[c] + beta[c];
        outf[(size_t)v * kC + c] = y;
        if (outb) outb[(size_t)v * kC + c] = (bf16_t)y;
    }
}

// ---------------------------------------------------------------------------
// Host orchestration
// ---------------------------------------------------------------------------
static void run_layer(const float* src_in, const int* inds, const unsigned char* mask,
                      const float* pos, void* const* P, float* out_f32,
                      char* ws, hipStream_t stream) {
    const float* in_w  = (const float*)P[0];
    const float* in_b  = (const float*)P[1];
    const float* out_w = (const float*)P[2];
    const float* out_b = (const float*)P[3];
    const float* n1_g  = (const float*)P[4];
    const float* n1_b  = (const float*)P[5];
    const float* l1_w  = (const float*)P[6];
    const float* l1_b  = (const float*)P[7];
    const float* l2_w  = (const float*)P[8];
    const float* l2_b  = (const float*)P[9];
    const float* n2_g  = (const float*)P[10];
    const float* n2_b  = (const float*)P[11];
    const float* n_g   = (const float*)P[12];
    const float* n_b   = (const float*)P[13];

    bf16_t* G_qk   = (bf16_t*)(ws + OFF_QK);
    bf16_t* G_feat = (bf16_t*)(ws + OFF_FEAT);
    bf16_t* G_qkv  = (bf16_t*)(ws + OFF_QKV);
    bf16_t* G_attn = G_qk;                       // alias: qk dead after QKV GEMM
    float*  F_tmp  = (float*)(ws + OFF_TMP);
    float*  F_x    = (float*)(ws + OFF_X);
    bf16_t* G_x    = G_feat;                     // alias: feat dead after QKV GEMM
    bf16_t* G_ffn1 = (bf16_t*)(ws + OFF_FFN1);
    float*  F_u    = (float*)(ws + OFF_QKV);     // alias: qkv dead after attention
    bf16_t* Win  = (bf16_t*)(ws + OFF_W);
    bf16_t* Wout = Win  + 3 * kC * kC;
    bf16_t* W1   = Wout + kC * kC;
    bf16_t* W2   = W1   + (size_t)kFF * kC;

    auto cvt = [&](const float* s, bf16_t* d, int n) {
        f2bf_kernel<<<(n + 255) / 256, 256, 0, stream>>>(s, d, n);
    };
    cvt(in_w,  Win,  3 * kC * kC);
    cvt(out_w, Wout, kC * kC);
    cvt(l1_w,  W1,   kFF * kC);
    cvt(l2_w,  W2,   kC * kFF);

    gather_kernel<<<kN, 256, 0, stream>>>(src_in, pos, inds, G_qk, G_feat);

    // QKV projection: Q,K from qk; V from feat. Output interleaved (M, 3C), ldc=1536.
    gemm_bf16_kernel<<<dim3(1024 / 128, kN / 128), 128, 0, stream>>>(
        G_qk, kC, Win, kC, in_b, nullptr, G_qkv, 3 * kC, kC, 0);
    gemm_bf16_kernel<<<dim3(kC / 128, kN / 128), 128, 0, stream>>>(
        G_feat, kC, Win + (size_t)2 * kC * kC, kC, in_b + 2 * kC,
        nullptr, G_qkv + 2 * kC, 3 * kC, kC, 0);

    attn_kernel<<<kS * kH, 32, 0, stream>>>(G_qkv, mask, G_attn);

    // out projection -> f32 tmp (set order)
    gemm_bf16_kernel<<<dim3(kC / 128, kN / 128), 128, 0, stream>>>(
        G_attn, kC, Wout, kC, out_b, F_tmp, nullptr, kC, kC, 0);

    // x = LN(src + scatter(tmp)) ; also bf16 mirror for FFN
    add_ln_kernel<<<kN, 256, 0, stream>>>(src_in, F_tmp, inds, n1_g, n1_b, F_x, G_x);

    // FFN
    gemm_bf16_kernel<<<dim3(kFF / 128, kN / 128), 128, 0, stream>>>(
        G_x, kC, W1, kC, l1_b, nullptr, G_ffn1, kFF, kC, 1);
    gemm_bf16_kernel<<<dim3(kC / 128, kN / 128), 128, 0, stream>>>(
        G_ffn1, kFF, W2, kFF, l2_b, F_tmp, nullptr, kC, kFF, 0);

    // y = LN(x + h) ; layer_out = LN(y + src)
    add_ln_kernel<<<kN, 256, 0, stream>>>(F_x, F_tmp, nullptr, n2_g, n2_b, F_u, nullptr);
    add_ln_kernel<<<kN, 256, 0, stream>>>(F_u, src_in, nullptr, n_g, n_b, out_f32, nullptr);
}

extern "C" void kernel_launch(void* const* d_in, const int* in_sizes, int n_in,
                              void* d_out, int out_size, void* d_ws, size_t ws_size,
                              hipStream_t stream) {
    const float* src          = (const float*)d_in[0];
    const int*   inds0        = (const int*)d_in[1];
    const int*   inds1        = (const int*)d_in[2];
    const unsigned char* msk0 = (const unsigned char*)d_in[3];
    const unsigned char* msk1 = (const unsigned char*)d_in[4];
    const float* pos0         = (const float*)d_in[5];
    const float* pos1         = (const float*)d_in[6];
    void* const* P0 = d_in + 7;
    void* const* P1 = d_in + 21;

    char*  ws  = (char*)d_ws;
    float* F_y = (float*)(ws + OFF_Y);

    run_layer(src, inds0, msk0, pos0, P0, F_y, ws, stream);
    run_layer(F_y, inds1, msk1, pos1, P1, (float*)d_out, ws, stream);
}
